// SignedMultiHopPropagation_26603027432197
// MI455X (gfx1250) — matrix operand order repaired and checked by reading, not compiled, not run on hardware
//
#include <hip/hip_runtime.h>
#include <math.h>

#define N_NODES 50000
#define D 256
#define HOPS 3
#define WROW 264       // bf16 elements per LDS weight row: 528B, 16B-aligned, bank-spread
#define EPB 16         // edges per block in spmm

typedef __attribute__((ext_vector_type(16))) __bf16 v16bf;
typedef __attribute__((ext_vector_type(8)))  __bf16 v8bf;
typedef __attribute__((ext_vector_type(8)))  float  v8f;

// ---------------------------------------------------------------- utilities
__global__ void zero_kernel(float* __restrict__ p, long long n) {
  long long i = (long long)blockIdx.x * blockDim.x + threadIdx.x;
  if (i < n) p[i] = 0.0f;
}

// Wc_pos[h] = wfuse_w[h][:, :256] @ wpos_w[h]   (and same for neg with [:,256:])
// grid = HOPS*2*256 blocks (h, sign, out-row i), 256 threads (out-col j)
__global__ void fold_weights_kernel(const float* __restrict__ wfuse_w,
                                    const float* __restrict__ wpos_w,
                                    const float* __restrict__ wneg_w,
                                    float* __restrict__ wc_pos,
                                    float* __restrict__ wc_neg) {
  int b    = blockIdx.x;
  int h    = b >> 9;         // / 512
  int rem  = b & 511;
  int sign = rem >> 8;
  int i    = rem & 255;
  int j    = threadIdx.x;
  const float* wf = wfuse_w + ((long long)(h * 256 + i)) * 512 + sign * 256;
  const float* ws = (sign ? wneg_w : wpos_w) + (long long)h * 256 * 256;
  float acc = 0.0f;
  for (int k = 0; k < 256; ++k)
    acc += wf[k] * ws[k * 256 + j];
  float* outp = sign ? wc_neg : wc_pos;
  outp[(long long)h * 256 * 256 + i * 256 + j] = acc;
}

// bc[h] = wfuse_b[h] + Wf1[h]@wpos_b[h] + Wf2[h]@wneg_b[h]
__global__ void fold_bias_kernel(const float* __restrict__ wfuse_w,
                                 const float* __restrict__ wfuse_b,
                                 const float* __restrict__ wpos_b,
                                 const float* __restrict__ wneg_b,
                                 float* __restrict__ bc) {
  int h = blockIdx.x, i = threadIdx.x;
  const float* wf = wfuse_w + ((long long)(h * 256 + i)) * 512;
  float acc = wfuse_b[h * 256 + i];
  for (int k = 0; k < 256; ++k)
    acc += wf[k] * wpos_b[h * 256 + k] + wf[256 + k] * wneg_b[h * 256 + k];
  bc[h * 256 + i] = acc;
}

// ------------------------------------------------------------------- SpMM
// acc[row] += val * z[col] ; 1 thread per feature column, EPB edges per block
__global__ __launch_bounds__(256) void spmm_kernel(const int* __restrict__ idx,
                                                   const float* __restrict__ val,
                                                   const float* __restrict__ z,
                                                   float* __restrict__ acc, int nE) {
  __shared__ int   s_row[EPB];
  __shared__ int   s_col[EPB];
  __shared__ float s_val[EPB];
  int t = threadIdx.x;
  int base = blockIdx.x * EPB;
  if (t < EPB) {
    int e = base + t;
    if (e < nE) { s_row[t] = idx[e]; s_col[t] = idx[nE + e]; s_val[t] = val[e]; }
    else        { s_row[t] = 0;      s_col[t] = 0;           s_val[t] = 0.0f;   }
  }
  __syncthreads();
#pragma unroll
  for (int i = 0; i < EPB; ++i) {
    float v = s_val[i] * z[(long long)s_col[i] * D + t];
    unsafeAtomicAdd(&acc[(long long)s_row[i] * D + t], v);  // global_atomic_add_f32
  }
}

// ------------------------------------------------------------------- GEMM
// A fragment (bf16 16x32): lane L -> row L&15; K chunks [(L>>4)*8,+8) and [16+(L>>4)*8,+8)
__device__ __forceinline__ v16bf load_a_frag(const float* __restrict__ arow, int k0) {
  const float* p = arow + k0;
  float tmp[16];
  *(float4*)&tmp[0]  = ((const float4*)p)[0];
  *(float4*)&tmp[4]  = ((const float4*)p)[1];
  *(float4*)&tmp[8]  = ((const float4*)p)[4];       // +16 floats (K+16)
  *(float4*)&tmp[12] = ((const float4*)p)[5];
  v16bf f;
#pragma unroll
  for (int i = 0; i < 8; ++i) { f[i] = (__bf16)tmp[i]; f[8 + i] = (__bf16)tmp[8 + i]; }
  return f;
}

// B fragment (bf16 32x16): lane L -> column L&15; K chunks [(L>>4)*8,+8) and [16+(L>>4)*8,+8)
__device__ __forceinline__ v16bf load_b_frag(const __bf16* __restrict__ brow, int k0) {
  const __bf16* p = brow + k0;                      // brow already includes lane n & k-half
  v8bf lo = *(const v8bf*)p;
  v8bf hi = *(const v8bf*)(p + 16);
  v16bf f;
#pragma unroll
  for (int i = 0; i < 8; ++i) { f[i] = lo[i]; f[8 + i] = hi[i]; }
  return f;
}

// C[M,256] = act( A1@W1^T (+ A2@W2^T) + bias ), W row-major [256 out, 256 in].
// 8 waves: 4(M) x 2(N); each wave: 16 rows x 128 cols = 8 wmma accumulators.
// Whole 256x256 W staged once per pass as bf16 in LDS; K loop is barrier-free.
__global__ __launch_bounds__(256) void gemm_kernel(
    const float* __restrict__ A1, const float* __restrict__ W1,
    const float* __restrict__ A2, const float* __restrict__ W2,
    const float* __restrict__ bias, float* __restrict__ C,
    int M, int act_tanh) {
  __shared__ __bf16 lw[256 * WROW];                 // 132 KB bf16 weight matrix
  int tid  = threadIdx.x;
  int wid  = tid >> 5, lane = tid & 31;
  int wave_m = wid >> 1, wave_n = wid & 1;
  int m0  = blockIdx.x * 64 + wave_m * 16;
  int nb0 = wave_n * 128;

  // per-lane fixed coordinates
  int arow_i = m0 + (lane & 15);
  if (arow_i >= M) arow_i = M - 1;                  // clamped; masked at store
  const int a_koff = (lane >> 4) * 8;               // A frag K sub-offset
  // B frag: lane -> column (L&15), K sub-offset (L>>4)*8
  const __bf16* bbase = &lw[(nb0 + (lane & 15)) * WROW + (lane >> 4) * 8];

  v8f acc[8];
#pragma unroll
  for (int i = 0; i < 8; ++i) acc[i] = (v8f){0, 0, 0, 0, 0, 0, 0, 0};

  const int npass = (A2 != nullptr) ? 2 : 1;
  for (int pass = 0; pass < npass; ++pass) {
    const float* A = pass ? A2 : A1;
    const float* W = pass ? W2 : W1;
    const float* arow = A + (long long)arow_i * D + a_koff;

    __syncthreads();                                // protect LDS from previous pass readers
    {                                               // stage full W as bf16: thread t owns row t
      const float4* wq = (const float4*)(W + (long long)tid * D);
      __bf16* drow = &lw[tid * WROW];
#pragma unroll 8
      for (int c = 0; c < 64; ++c) {
        float4 v = wq[c];
        drow[c * 4 + 0] = (__bf16)v.x; drow[c * 4 + 1] = (__bf16)v.y;
        drow[c * 4 + 2] = (__bf16)v.z; drow[c * 4 + 3] = (__bf16)v.w;
      }
    }
    __syncthreads();

    // software-pipelined K loop: prefetch next A frag (global) and next B frag (LDS)
    v16bf afrag = load_a_frag(arow, 0);
#pragma unroll
    for (int k0 = 0; k0 < D; k0 += 32) {
      v16bf anext = afrag;
      if (k0 + 32 < D) anext = load_a_frag(arow, k0 + 32);
      v16bf bfrag = load_b_frag(bbase, k0);
#pragma unroll
      for (int nt = 0; nt < 8; ++nt) {
        v16bf bnext = bfrag;
        if (nt < 7) bnext = load_b_frag(bbase + (nt + 1) * 16 * WROW, k0);
        acc[nt] = __builtin_amdgcn_wmma_f32_16x16x32_bf16(
            false, afrag, false, bfrag, (short)0, acc[nt], false, false);
        bfrag = bnext;
      }
      afrag = anext;
    }
  }

  // epilogue: D layout — VGPR v, lane L -> (m = v + 8*(L>>4), n = L&15)
  int cn   = lane & 15;
  int rsel = lane >> 4;
#pragma unroll
  for (int nt = 0; nt < 8; ++nt) {
    int n = nb0 + nt * 16 + cn;
    float b = bias ? bias[n] : 0.0f;
#pragma unroll
    for (int v = 0; v < 8; ++v) {
      int m = m0 + v + 8 * rsel;
      if (m < M) {
        float r = acc[nt][v] + b;
        if (act_tanh) r = tanhf(r);
        C[(long long)m * D + n] = r;
      }
    }
  }
}

// ------------------------------------------------------------------ driver
extern "C" void kernel_launch(void* const* d_in, const int* in_sizes, int n_in,
                              void* d_out, int out_size, void* d_ws, size_t ws_size,
                              hipStream_t stream) {
  const float* x       = (const float*)d_in[0];
  const int*   pos_idx = (const int*)  d_in[1];
  const float* pos_val = (const float*)d_in[2];
  const int*   neg_idx = (const int*)  d_in[3];
  const float* neg_val = (const float*)d_in[4];
  const float* in_w    = (const float*)d_in[5];
  const float* in_b    = (const float*)d_in[6];
  const float* wpos_w  = (const float*)d_in[7];
  const float* wpos_b  = (const float*)d_in[8];
  const float* wneg_w  = (const float*)d_in[9];
  const float* wneg_b  = (const float*)d_in[10];
  const float* wfuse_w = (const float*)d_in[11];
  const float* wfuse_b = (const float*)d_in[12];
  float* out = (float*)d_out;

  const int E = in_sizes[2];                 // edges per sign
  const long long slab = (long long)N_NODES * D;

  // workspace layout (fp32): acc_pos | acc_neg | Wc_pos | Wc_neg | bc  (~104 MB)
  float* acc_pos = (float*)d_ws;
  float* acc_neg = acc_pos + slab;
  float* wc_pos  = acc_neg + slab;
  float* wc_neg  = wc_pos + (long long)HOPS * D * D;
  float* bc      = wc_neg + (long long)HOPS * D * D;

  // fold per-hop linear layers: 3 GEMMs/hop -> 1 fused two-input GEMM/hop
  fold_weights_kernel<<<HOPS * 2 * D, 256, 0, stream>>>(wfuse_w, wpos_w, wneg_w, wc_pos, wc_neg);
  fold_bias_kernel<<<HOPS, 256, 0, stream>>>(wfuse_w, wfuse_b, wpos_b, wneg_b, bc);

  const int gemm_blocks = (N_NODES + 63) / 64;

  // z0 = tanh(x @ in_w^T + in_b) -> out slab 0
  gemm_kernel<<<gemm_blocks, 256, 0, stream>>>(x, in_w, nullptr, nullptr, in_b,
                                               out, N_NODES, 1);

  const int spmm_blocks = (E + EPB - 1) / EPB;
  const int zero_blocks = (int)((2 * slab + 255) / 256);

  for (int h = 0; h < HOPS; ++h) {
    const float* z = out + (long long)h * slab;
    zero_kernel<<<zero_blocks, 256, 0, stream>>>(acc_pos, 2 * slab);  // both accs contiguous
    spmm_kernel<<<spmm_blocks, 256, 0, stream>>>(pos_idx, pos_val, z, acc_pos, E);
    spmm_kernel<<<spmm_blocks, 256, 0, stream>>>(neg_idx, neg_val, z, acc_neg, E);
    // z_{h+1} = tanh(acc_pos@Wc_pos^T + acc_neg@Wc_neg^T + bc) -> out slab h+1
    gemm_kernel<<<gemm_blocks, 256, 0, stream>>>(acc_pos, wc_pos + (long long)h * D * D,
                                                 acc_neg, wc_neg + (long long)h * D * D,
                                                 bc + h * D,
                                                 out + (long long)(h + 1) * slab,
                                                 N_NODES, 1);
  }
}
